// ChebNetII_89326729822512
// MI455X (gfx1250) — matrix operand order repaired and checked by reading, not compiled, not run on hardware
//
#include <hip/hip_runtime.h>
#include <hip/hip_bf16.h>

// ---------------------------------------------------------------------------
// ChebNetII forward for MI455X (gfx1250, wave32, WMMA)
//   x[N,512] -> relu(x@W1+b1) -> z = h@W2+b2 -> K=10 Chebyshev props -> logsoftmax
// ---------------------------------------------------------------------------

#define F_IN 512
#define HID  256
#define CDIM 64
#define KCH  10          // Chebyshev order K
#define BM   128         // GEMM block tile rows
#define BK   32          // GEMM k-tile (matches 16x16x32 wmma)

typedef __attribute__((ext_vector_type(16))) __bf16 v16bf;
typedef __attribute__((ext_vector_type(8)))  __bf16 v8bf;
typedef __attribute__((ext_vector_type(8)))  float  v8f;

union FragBf { v16bf v; v8bf h[2]; };

// ---------------- small elementwise / setup kernels ------------------------

__global__ void zero_f32_kernel(float* __restrict__ p, long long n) {
    long long i = (long long)blockIdx.x * blockDim.x + threadIdx.x;
    long long stride = (long long)gridDim.x * blockDim.x;
    for (; i < n; i += stride) p[i] = 0.0f;
}

// coe[i] = 2/(K+1) * sum_j relu(temp[j]) * T_i(x_j),  T_i(cos t) = cos(i*t)
__global__ void coe_kernel(const float* __restrict__ temp, float* __restrict__ coe) {
    int i = threadIdx.x;
    if (i > KCH) return;
    const float pi = 3.14159265358979323846f;
    float acc = 0.0f;
    for (int j = 0; j <= KCH; ++j) {
        float th = (float)(KCH - j) + 0.5f;
        th = th * pi / (float)(KCH + 1);
        float t = temp[j];
        t = t > 0.0f ? t : 0.0f;
        acc += t * cosf((float)i * th);
    }
    coe[i] = (2.0f / (float)(KCH + 1)) * acc;
}

__global__ void degree_kernel(const long long* __restrict__ src,
                              float* __restrict__ deg, int E) {
    int e = blockIdx.x * blockDim.x + threadIdx.x;
    if (e >= E) return;
    unsafeAtomicAdd(&deg[src[e]], 1.0f);
}

__global__ void dinv_kernel(float* __restrict__ deg_dinv, int n) {
    int i = blockIdx.x * blockDim.x + threadIdx.x;
    if (i >= n) return;
    float d = deg_dinv[i];
    deg_dinv[i] = (d > 0.0f) ? rsqrtf(fmaxf(d, 1.0f)) : 0.0f;
}

__global__ void norm_kernel(const long long* __restrict__ src,
                            const long long* __restrict__ dst,
                            const float* __restrict__ dinv,
                            float* __restrict__ norm, int E) {
    int e = blockIdx.x * blockDim.x + threadIdx.x;
    if (e >= E) return;
    norm[e] = -(dinv[src[e]] * dinv[dst[e]]);
}

// ---------------- weight pre-swizzle into WMMA B-fragment layout -----------
// B fragment (32x16, bf16, wave32): lane L (L<16): N = L,  K = j      (j=0..15)
//                                   lane L (>=16): N = L-16, K = 16+j
// Stored linearly: Wswz[((kt*NT + nt)*32 + lane)*16 + j]
__global__ void swizzle_w_kernel(const float* __restrict__ W,
                                 __bf16* __restrict__ Wswz,
                                 int Kdim, int Ndim) {
    int total = Kdim * Ndim;
    int NT = Ndim >> 4;
    for (int t = blockIdx.x * blockDim.x + threadIdx.x; t < total;
         t += gridDim.x * blockDim.x) {
        int j    = t & 15;
        int lane = (t >> 4) & 31;
        int frag = t >> 9;
        int kt   = frag / NT;
        int ntg  = frag % NT;
        int k = kt * BK + ((lane < 16) ? j : 16 + j);
        int n = ntg * 16 + (lane & 15);
        Wswz[t] = (__bf16)W[(long long)k * Ndim + n];
    }
}

// ---------------- bf16 WMMA GEMM:  C[M,Nd] = act(A[M,Kd] @ W + bias) -------
// grid = (ceil(M/BM), Nd/64), block = 256 (8 waves); wave w owns rows
// [w*16, w*16+16) of the block tile and 4 n-tiles (64 cols).
__global__ void __launch_bounds__(256)
gemm_bf16_wmma_kernel(const float* __restrict__ A,
                      const __bf16* __restrict__ Bswz,
                      const float* __restrict__ bias,
                      float* __restrict__ Cout,
                      int M, int Kdim, int Ndim, int do_relu) {
    __shared__ alignas(16) __bf16 As[BM * BK];      // 8 KB

    const int tid   = threadIdx.x;
    const int wave  = tid >> 5;
    const int lane  = tid & 31;
    const int lhalf = lane & 15;
    const bool hiL  = lane >= 16;
    const int m0    = blockIdx.x * BM;
    const int n0    = blockIdx.y * 64;
    const int NT    = Ndim >> 4;
    const int ntg0  = n0 >> 4;
    const int KT    = Kdim >> 5;

    // staging: thread -> (row tid/2, 16-float chunk (tid&1)*16)
    const int srow = tid >> 1;
    const int scol = (tid & 1) << 4;

    v8f acc[4] = {};

    for (int kt = 0; kt < KT; ++kt) {
        // ---- stage A tile (BM x BK) f32 -> bf16 in LDS -------------------
        int gr = m0 + srow;
        if (gr >= M) gr = M - 1;                    // clamp; padded rows never stored
        const float* ap = A + (long long)gr * Kdim + kt * BK + scol;
        float4 f0 = ((const float4*)ap)[0];
        float4 f1 = ((const float4*)ap)[1];
        float4 f2 = ((const float4*)ap)[2];
        float4 f3 = ((const float4*)ap)[3];
        v8bf lo8, hi8;
        lo8[0]=(__bf16)f0.x; lo8[1]=(__bf16)f0.y; lo8[2]=(__bf16)f0.z; lo8[3]=(__bf16)f0.w;
        lo8[4]=(__bf16)f1.x; lo8[5]=(__bf16)f1.y; lo8[6]=(__bf16)f1.z; lo8[7]=(__bf16)f1.w;
        hi8[0]=(__bf16)f2.x; hi8[1]=(__bf16)f2.y; hi8[2]=(__bf16)f2.z; hi8[3]=(__bf16)f2.w;
        hi8[4]=(__bf16)f3.x; hi8[5]=(__bf16)f3.y; hi8[6]=(__bf16)f3.z; hi8[7]=(__bf16)f3.w;
        __bf16* sp = &As[srow * BK + scol];
        *(v8bf*)sp       = lo8;
        *(v8bf*)(sp + 8) = hi8;
        __syncthreads();

        // ---- A fragment: 16x32 bf16 contiguous k order (matches Wswz) -----------------------
        // lane<16: M=lane, K = 0..15
        // lane>=16: M=lane-16, K = 16..31
        FragBf afrag;
        const __bf16* abase = &As[(wave * 16 + lhalf) * BK + (hiL ? 16 : 0)];
        afrag.h[0] = *(const v8bf*)abase;
        afrag.h[1] = *(const v8bf*)(abase + 8);

        #pragma unroll
        for (int nt = 0; nt < 4; ++nt) {
            FragBf bfrag;
            const __bf16* bbase =
                Bswz + (((long long)kt * NT + (ntg0 + nt)) * 32 + lane) * 16;
            bfrag.h[0] = *(const v8bf*)bbase;
            bfrag.h[1] = *(const v8bf*)(bbase + 8);
            acc[nt] = __builtin_amdgcn_wmma_f32_16x16x32_bf16(
                false, afrag.v, false, bfrag.v, (short)0, acc[nt], false, false);
        }
        __syncthreads();
    }

    // ---- store C: lane<16 -> M=r, lane>=16 -> M=8+r, N = lane&15 ---------
    #pragma unroll
    for (int nt = 0; nt < 4; ++nt) {
        int n = n0 + nt * 16 + lhalf;
        float bv = bias[n];
        #pragma unroll
        for (int r = 0; r < 8; ++r) {
            int m = m0 + wave * 16 + (hiL ? 8 + r : r);
            if (m < M) {
                float v = acc[nt][r] + bv;
                if (do_relu) v = fmaxf(v, 0.0f);
                Cout[(long long)m * Ndim + n] = v;
            }
        }
    }
}

// ---------------- sparse propagation: out[dst] += norm * in[src] ----------
// 64 threads per edge (one per feature); grid-stride with prefetch of the
// next gathered source row (emits global_prefetch_b8 on gfx1250).
__global__ void prop_kernel(const long long* __restrict__ src,
                            const long long* __restrict__ dst,
                            const float* __restrict__ norm,
                            const float* __restrict__ in,
                            float* __restrict__ out, int E) {
    const int f = threadIdx.x & 63;
    const int epb = blockDim.x >> 6;                 // edges per block
    long long e = (long long)blockIdx.x * epb + (threadIdx.x >> 6);
    const long long stride = (long long)gridDim.x * epb;
    for (; e < E; e += stride) {
        long long en = e + stride;
        if (en < E) {
            long long sn = src[en];
            __builtin_prefetch(&in[sn * CDIM + f], 0, 0);
        }
        long long s = src[e];
        long long d = dst[e];
        float v = norm[e] * in[s * CDIM + f];
        unsafeAtomicAdd(&out[d * CDIM + f], v);
    }
}

// out = coe[0]/2 * t0 + coe[1] * t1
__global__ void out_init_kernel(const float* __restrict__ t0,
                                const float* __restrict__ t1,
                                float* __restrict__ out,
                                const float* __restrict__ coe, long long n) {
    float c0h = coe[0] * 0.5f;
    float c1  = coe[1];
    long long i = (long long)blockIdx.x * blockDim.x + threadIdx.x;
    long long stride = (long long)gridDim.x * blockDim.x;
    for (; i < n; i += stride) out[i] = c0h * t0[i] + c1 * t1[i];
}

// t2 holds prop(Tx1); compute Tx2 = 2*t2 - t0 (in place) and out += coe[i]*Tx2
__global__ void cheb_step_kernel(float* __restrict__ t2,
                                 const float* __restrict__ t0,
                                 float* __restrict__ out,
                                 const float* __restrict__ coe, int ci,
                                 long long n) {
    float c = coe[ci];
    long long i = (long long)blockIdx.x * blockDim.x + threadIdx.x;
    long long stride = (long long)gridDim.x * blockDim.x;
    for (; i < n; i += stride) {
        float v = 2.0f * t2[i] - t0[i];
        t2[i] = v;
        out[i] += c * v;
    }
}

// in-place log_softmax over rows of 64; one wave32 per row, 2 floats/lane
__global__ void logsoftmax64_kernel(float* __restrict__ out, int M) {
    int wave = threadIdx.x >> 5;
    int lane = threadIdx.x & 31;
    long long row = (long long)blockIdx.x * (blockDim.x >> 5) + wave;
    if (row >= M) return;
    float2 v = ((const float2*)(out + row * CDIM))[lane];
    float m = fmaxf(v.x, v.y);
    #pragma unroll
    for (int off = 16; off > 0; off >>= 1)
        m = fmaxf(m, __shfl_xor(m, off, 32));
    float s = expf(v.x - m) + expf(v.y - m);
    #pragma unroll
    for (int off = 16; off > 0; off >>= 1)
        s += __shfl_xor(s, off, 32);
    float l = m + logf(s);
    float2 r; r.x = v.x - l; r.y = v.y - l;
    ((float2*)(out + row * CDIM))[lane] = r;
}

// ---------------------------------------------------------------------------

extern "C" void kernel_launch(void* const* d_in, const int* in_sizes, int n_in,
                              void* d_out, int out_size, void* d_ws, size_t ws_size,
                              hipStream_t stream) {
    if (n_in < 7) return;
    const float*      x    = (const float*)d_in[0];
    const long long*  ei   = (const long long*)d_in[1];   // [2, E] int64
    const float*      W1   = (const float*)d_in[2];
    const float*      b1   = (const float*)d_in[3];
    const float*      W2   = (const float*)d_in[4];
    const float*      b2   = (const float*)d_in[5];
    const float*      temp = (const float*)d_in[6];
    float*            out  = (float*)d_out;

    const int N = in_sizes[0] / F_IN;
    const int E = in_sizes[1] / 2;
    const long long NC = (long long)N * CDIM;

    const long long* src = ei;
    const long long* dst = ei + E;

    // ---- carve workspace -------------------------------------------------
    char* p = (char*)d_ws;
    auto carve = [&](size_t bytes) -> void* {
        void* r = (void*)p;
        p += (bytes + 255) & ~(size_t)255;
        return r;
    };
    float*  coe  = (float*)carve(16 * sizeof(float));
    float*  dinv = (float*)carve((size_t)N * sizeof(float));   // deg -> dinv in place
    float*  nrm  = (float*)carve((size_t)E * sizeof(float));
    __bf16* W1s  = (__bf16*)carve((size_t)F_IN * HID * sizeof(__bf16));
    __bf16* W2s  = (__bf16*)carve((size_t)HID * CDIM * sizeof(__bf16));
    float*  h    = (float*)carve((size_t)N * HID * sizeof(float));
    float*  t0   = (float*)carve((size_t)NC * sizeof(float));  // z / Tx0
    float*  t1   = (float*)carve((size_t)NC * sizeof(float));
    float*  t2   = (float*)carve((size_t)NC * sizeof(float));
    (void)ws_size;

    const int TB = 256;

    // ---- Chebyshev coefficients -----------------------------------------
    coe_kernel<<<1, 32, 0, stream>>>(temp, coe);

    // ---- graph normalization --------------------------------------------
    zero_f32_kernel<<<1024, TB, 0, stream>>>(dinv, N);
    degree_kernel<<<(E + TB - 1) / TB, TB, 0, stream>>>(src, dinv, E);
    dinv_kernel<<<(N + TB - 1) / TB, TB, 0, stream>>>(dinv, N);
    norm_kernel<<<(E + TB - 1) / TB, TB, 0, stream>>>(src, dst, dinv, nrm, E);

    // ---- weight swizzle to WMMA fragment layout --------------------------
    swizzle_w_kernel<<<512, TB, 0, stream>>>(W1, W1s, F_IN, HID);
    swizzle_w_kernel<<<64,  TB, 0, stream>>>(W2, W2s, HID, CDIM);

    // ---- MLP: h = relu(x@W1+b1), z = h@W2+b2 (z in t0) -------------------
    {
        dim3 g1((N + BM - 1) / BM, HID / 64);
        gemm_bf16_wmma_kernel<<<g1, TB, 0, stream>>>(x, W1s, b1, h, N, F_IN, HID, 1);
        dim3 g2((N + BM - 1) / BM, CDIM / 64);
        gemm_bf16_wmma_kernel<<<g2, TB, 0, stream>>>(h, W2s, b2, t0, N, HID, CDIM, 0);
    }

    // ---- Chebyshev recurrence --------------------------------------------
    const int PROP_GRID = 4096;
    const int EW_GRID   = 2048;

    // Tx1 = prop(z)
    zero_f32_kernel<<<EW_GRID, TB, 0, stream>>>(t1, NC);
    prop_kernel<<<PROP_GRID, TB, 0, stream>>>(src, dst, nrm, t0, t1, E);
    // out = coe0/2 * Tx0 + coe1 * Tx1
    out_init_kernel<<<EW_GRID, TB, 0, stream>>>(t0, t1, out, coe, NC);

    float* Tx0 = t0;
    float* Tx1 = t1;
    float* Tx2 = t2;
    for (int i = 2; i <= KCH; ++i) {
        zero_f32_kernel<<<EW_GRID, TB, 0, stream>>>(Tx2, NC);
        prop_kernel<<<PROP_GRID, TB, 0, stream>>>(src, dst, nrm, Tx1, Tx2, E);
        cheb_step_kernel<<<EW_GRID, TB, 0, stream>>>(Tx2, Tx0, out, coe, i, NC);
        float* tmp = Tx0; Tx0 = Tx1; Tx1 = Tx2; Tx2 = tmp;
    }

    // ---- log_softmax in place --------------------------------------------
    logsoftmax64_kernel<<<(N + 7) / 8, TB, 0, stream>>>(out, N);
}
